// TransformerModel_71725953843748
// MI455X (gfx1250) — matrix operand order repaired and checked by reading, not compiled, not run on hardware
//
#include <hip/hip_runtime.h>
#include <hip/hip_bf16.h>

// ---------------- model dims ----------------
constexpr int Bn = 4, Sn = 512, Dn = 768, Hn = 12, HDn = 64, Ln = 12, Fn = 3072;
constexpr int Mn = Bn * Sn;          // 2048 rows
constexpr float LN_EPS = 1e-3f;

// ---------------- vector types ----------------
typedef __attribute__((ext_vector_type(16))) __bf16          v16bf;
typedef __attribute__((ext_vector_type(8)))  float           v8f;
typedef __attribute__((ext_vector_type(8)))  unsigned short  v8us;
typedef __attribute__((ext_vector_type(16))) unsigned short  v16us;

// pack two f32 -> bf16x2 (round-to-nearest via +0x8000, packed with v_perm_b32)
__device__ __forceinline__ unsigned pk_bf16(float x, float y) {
#if __has_builtin(__builtin_amdgcn_perm)
    const unsigned ux = __builtin_bit_cast(unsigned, x) + 0x8000u;
    const unsigned uy = __builtin_bit_cast(unsigned, y) + 0x8000u;
    // result bytes: [1:0] = ux[3:2], [3:2] = uy[3:2]
    return __builtin_amdgcn_perm(uy, ux, 0x07060302u);
#else
    unsigned ux = __builtin_bit_cast(unsigned, x);
    unsigned uy = __builtin_bit_cast(unsigned, y);
    ux += 0x7FFFu + ((ux >> 16) & 1u);
    uy += 0x7FFFu + ((uy >> 16) & 1u);
    return (ux >> 16) | (uy & 0xFFFF0000u);
#endif
}

// float4 -> packed 4x bf16 (two dwords)
__device__ __forceinline__ uint2 cvt4(float4 f) {
    uint2 u;
    u.x = pk_bf16(f.x, f.y);
    u.y = pk_bf16(f.z, f.w);
    return u;
}

__device__ __forceinline__ v16bf mk16(v8us lo, v8us hi) {
    v16us t;
#pragma unroll
    for (int e = 0; e < 8; ++e) { t[e] = lo[e]; t[e + 8] = hi[e]; }
    return __builtin_bit_cast(v16bf, t);
}

// ---------------- embedding + sinusoidal positions ----------------
__global__ __launch_bounds__(256)
void embed_kernel(const int* __restrict__ ids, const int* __restrict__ segs,
                  const float* __restrict__ tok, const float* __restrict__ seg,
                  float* __restrict__ x) {
    const int idx = blockIdx.x * 256 + threadIdx.x;       // over M*D
    const int m = idx / Dn, d = idx % Dn;
    const int s = m % Sn;
    float pe;
    if (s > 0 && (d & 1)) {
        pe = 0.0f;                                        // pe[1:,1::2] = 0 quirk
    } else {
        const int j = (d < Dn / 2) ? d : d - Dn / 2;
        const float div = __expf(-(float)(2 * j) * (9.210340371976184f / (float)Dn));
        const float ang = (float)s * div;
        pe = (d < Dn / 2) ? __sinf(ang) : __cosf(ang);
    }
    const float t = tok[(size_t)ids[m] * Dn + d] * 27.712812921102035f; // sqrt(768)
    x[idx] = t + pe + seg[(size_t)segs[m] * Dn + d];
}

// ---------------- generic bf16-WMMA tiled GEMM, double-buffered ----------------
// C[M,N] = alpha * A[M,K] @ B (+ bias) (+ res), optional relu -- all compile-time.
// TRANSB=false: B is [K,N] row-major.  TRANSB=true: B source is [N,K] row-major.
// blockIdx.z batches (b,h): offsets = (z/Hdiv)*s?b + (z%Hdiv)*s?h
template <int TM, int TN, bool TRANSB, bool HAS_BIAS, bool HAS_RES, bool DO_RELU>
__global__ __launch_bounds__(256)
void gemm_wmma(const float* __restrict__ A, int lda, long long sAb, long long sAh,
               const float* __restrict__ Bm, int ldb, long long sBb, long long sBh,
               float* __restrict__ C, int ldc, long long sCb, long long sCh,
               const float* __restrict__ bias, const float* __restrict__ res,
               int K, int Hdiv, float alpha) {
    constexpr int WCOLS = TN / 32;
    constexpr int ARE = TM / 32;     // float4 regs per thread for A tile
    constexpr int BRE = TN / 32;     // float4 regs per thread for B tile
    __shared__ __align__(16) unsigned short Alds[2][TM][40];
    __shared__ __align__(16) unsigned short Blds[2][TN][40];

    const int tid = threadIdx.x;
    const int wave = tid >> 5, lane = tid & 31;
    const int l16 = lane & 15, g = lane >> 4;
    const int wm = wave / WCOLS, wn = wave % WCOLS;
    const int tileM = blockIdx.y * TM, tileN = blockIdx.x * TN;
    const long long zb = blockIdx.z / Hdiv, zh = blockIdx.z % Hdiv;
    const long long offA = zb * sAb + zh * sAh;
    const long long offB = zb * sBb + zh * sBh;
    const long long offC = zb * sCb + zh * sCh;

    v8f acc[2][2] = {{{}, {}}, {{}, {}}};
    float4 aReg[ARE], bReg[BRE];

    // ---- register-staged tile fetch / LDS commit ----
    auto loadA = [&](int k0) {
#pragma unroll
        for (int it = 0; it < ARE; ++it) {
            const int idx = tid + it * 256;
            const int row = idx >> 3, c4 = (idx & 7) << 2;
            aReg[it] = *(const float4*)(A + offA + (long long)(tileM + row) * lda + (k0 + c4));
        }
    };
    auto storeA = [&](int p) {
#pragma unroll
        for (int it = 0; it < ARE; ++it) {
            const int idx = tid + it * 256;
            const int row = idx >> 3, c4 = (idx & 7) << 2;
            *(uint2*)&Alds[p][row][c4] = cvt4(aReg[it]);
        }
    };
    auto loadB = [&](int k0) {
#pragma unroll
        for (int it = 0; it < BRE; ++it) {
            const int idx = tid + it * 256;
            if (TRANSB) {
                const int n = idx >> 3, c4 = (idx & 7) << 2;
                bReg[it] = *(const float4*)(Bm + offB + (long long)(tileN + n) * ldb + (k0 + c4));
            } else {
                // column strip: 4 K-values at fixed n (coalesced b32 loads),
                // so the transposed LDS commit is one contiguous 8-byte store
                const int n = idx % TN, kq = idx / TN;
                const float* src = Bm + offB + (long long)(k0 + kq * 4) * ldb + (tileN + n);
                bReg[it].x = src[0];
                bReg[it].y = src[ldb];
                bReg[it].z = src[2 * (long long)ldb];
                bReg[it].w = src[3 * (long long)ldb];
            }
        }
    };
    auto storeB = [&](int p) {
#pragma unroll
        for (int it = 0; it < BRE; ++it) {
            const int idx = tid + it * 256;
            if (TRANSB) {
                const int n = idx >> 3, c4 = (idx & 7) << 2;
                *(uint2*)&Blds[p][n][c4] = cvt4(bReg[it]);
            } else {
                const int n = idx % TN, kq = idx / TN;
                *(uint2*)&Blds[p][n][kq * 4] = cvt4(bReg[it]);
            }
        }
    };

    // ---- software pipeline: prologue ----
    loadA(0); loadB(0);
    storeA(0); storeB(0);

    int p = 0;
    for (int k0 = 0; k0 < K; k0 += 32) {
        __syncthreads();
        const bool more = (k0 + 32) < K;
        if (more) { loadA(k0 + 32); loadB(k0 + 32); }   // issue next tile early

        // ---- compute current buffer (fragments per ISA 7.12.2 layouts) ----
        v16bf af[2], bfv[2];
#pragma unroll
        for (int i = 0; i < 2; ++i) {
            const int row = wm * 32 + i * 16 + l16;
            af[i] = mk16(*(const v8us*)&Alds[p][row][g * 8],
                         *(const v8us*)&Alds[p][row][16 + g * 8]);
        }
#pragma unroll
        for (int j = 0; j < 2; ++j) {
            const int n = wn * 32 + j * 16 + l16;
            bfv[j] = mk16(*(const v8us*)&Blds[p][n][g * 8],
                          *(const v8us*)&Blds[p][n][16 + g * 8]);
        }
#pragma unroll
        for (int i = 0; i < 2; ++i)
#pragma unroll
            for (int j = 0; j < 2; ++j)
                acc[i][j] = __builtin_amdgcn_wmma_f32_16x16x32_bf16(
                    false, af[i], false, bfv[j], (short)0, acc[i][j], false, false);

        if (more) { storeA(p ^ 1); storeB(p ^ 1); }     // commit next tile
        p ^= 1;
    }

    // ---- epilogue: alpha, bias, residual, relu (all compile-time) ----
#pragma unroll
    for (int i = 0; i < 2; ++i)
#pragma unroll
        for (int j = 0; j < 2; ++j) {
            const int col = tileN + wn * 32 + j * 16 + l16;
            const float bv = HAS_BIAS ? bias[col] : 0.0f;
#pragma unroll
            for (int r = 0; r < 8; ++r) {
                const int row = tileM + wm * 32 + i * 16 + g * 8 + r;
                float v = acc[i][j][r] * alpha + bv;
                if (HAS_RES) v += res[offC + (long long)row * ldc + col];
                if (DO_RELU) v = fmaxf(v, 0.0f);
                C[offC + (long long)row * ldc + col] = v;
            }
        }
}

// ---------------- row softmax over 512 (in place) ----------------
__global__ __launch_bounds__(256)
void softmax512(float* __restrict__ p) {
    __shared__ float red[256];
    float* row = p + (size_t)blockIdx.x * 512;
    const int t = threadIdx.x;
    const float a = row[t], b = row[t + 256];
    red[t] = fmaxf(a, b);
    __syncthreads();
    for (int s = 128; s > 0; s >>= 1) {
        if (t < s) red[t] = fmaxf(red[t], red[t + s]);
        __syncthreads();
    }
    const float m = red[0];
    __syncthreads();
    const float ea = __expf(a - m), eb = __expf(b - m);
    red[t] = ea + eb;
    __syncthreads();
    for (int s = 128; s > 0; s >>= 1) {
        if (t < s) red[t] += red[t + s];
        __syncthreads();
    }
    const float inv = 1.0f / red[0];
    row[t] = ea * inv;
    row[t + 256] = eb * inv;
}

// ---------------- LayerNorm over D=768 (optional fused residual) ----------------
__global__ __launch_bounds__(256)
void ln_kernel(const float* __restrict__ a, const float* __restrict__ bres,
               const float* __restrict__ gam, const float* __restrict__ bet,
               float* __restrict__ out) {
    __shared__ float red[256];
    const int row = blockIdx.x, t = threadIdx.x;
    const float* ar = a + (size_t)row * Dn;
    float x0 = ar[t], x1 = ar[t + 256], x2 = ar[t + 512];
    if (bres) {
        const float* br = bres + (size_t)row * Dn;
        x0 += br[t]; x1 += br[t + 256]; x2 += br[t + 512];
    }
    red[t] = x0 + x1 + x2;
    __syncthreads();
    for (int s = 128; s > 0; s >>= 1) {
        if (t < s) red[t] += red[t + s];
        __syncthreads();
    }
    const float mean = red[0] * (1.0f / (float)Dn);
    __syncthreads();
    const float d0 = x0 - mean, d1 = x1 - mean, d2 = x2 - mean;
    red[t] = d0 * d0 + d1 * d1 + d2 * d2;
    __syncthreads();
    for (int s = 128; s > 0; s >>= 1) {
        if (t < s) red[t] += red[t + s];
        __syncthreads();
    }
    const float rstd = rsqrtf(red[0] * (1.0f / (float)Dn) + LN_EPS);
    float* o = out + (size_t)row * Dn;
    o[t]       = d0 * rstd * gam[t]       + bet[t];
    o[t + 256] = d1 * rstd * gam[t + 256] + bet[t + 256];
    o[t + 512] = d2 * rstd * gam[t + 512] + bet[t + 512];
}

// ---------------- host orchestration ----------------
extern "C" void kernel_launch(void* const* d_in, const int* in_sizes, int n_in,
                              void* d_out, int out_size, void* d_ws, size_t ws_size,
                              hipStream_t stream) {
    (void)in_sizes; (void)n_in; (void)out_size; (void)ws_size;

    const int*   ids     = (const int*)d_in[0];
    const int*   segs    = (const int*)d_in[1];
    const float* tok_emb = (const float*)d_in[2];
    const float* seg_emb = (const float*)d_in[3];
    const float* Wq = (const float*)d_in[4],  *bq = (const float*)d_in[5];
    const float* Wk = (const float*)d_in[6],  *bk = (const float*)d_in[7];
    const float* Wv = (const float*)d_in[8],  *bv = (const float*)d_in[9];
    const float* Wo = (const float*)d_in[10], *bo = (const float*)d_in[11];
    const float* W1 = (const float*)d_in[12], *b1 = (const float*)d_in[13];
    const float* W2 = (const float*)d_in[14], *b2 = (const float*)d_in[15];
    const float* ln_ffn_g = (const float*)d_in[16], *ln_ffn_b = (const float*)d_in[17];
    const float* ln1_g = (const float*)d_in[18], *ln1_b = (const float*)d_in[19];
    const float* ln2_g = (const float*)d_in[20], *ln2_b = (const float*)d_in[21];

    float* outp  = (float*)d_out;                       // [B,S,D]
    float* probs = outp + (size_t)Mn * Dn;              // [L,B,H,S,S]

    const size_t MD = (size_t)Mn * Dn;
    float* xe     = (float*)d_ws;
    float* h      = xe + MD;
    float* q      = h + MD;
    float* k      = q + MD;
    float* v      = k + MD;
    float* ctx    = v + MD;
    float* attln  = ctx + MD;
    float* hidden = attln + MD;                          // [M,F]
    float* ffn    = hidden + (size_t)Mn * Fn;

    // 1) embeddings
    embed_kernel<<<(Mn * Dn) / 256, 256, 0, stream>>>(ids, segs, tok_emb, seg_emb, xe);

    const long long sQ_b = (long long)Sn * Dn, sQ_h = HDn;           // per-(b,h) views of [M,D]
    const long long sP_b = (long long)Hn * Sn * Sn, sP_h = (long long)Sn * Sn;

    // 2) stacked attention layers (scan: h_{l+1} = attn_l(h_l))
    for (int l = 0; l < Ln; ++l) {
        const float* in_h = (l == 0) ? xe : h;
        const float* Wq_l = Wq + (size_t)l * Dn * Dn, *bq_l = bq + (size_t)l * Dn;
        const float* Wk_l = Wk + (size_t)l * Dn * Dn, *bk_l = bk + (size_t)l * Dn;
        const float* Wv_l = Wv + (size_t)l * Dn * Dn, *bv_l = bv + (size_t)l * Dn;
        const float* Wo_l = Wo + (size_t)l * Dn * Dn, *bo_l = bo + (size_t)l * Dn;
        float* probs_l = probs + (size_t)l * Bn * Hn * Sn * Sn;

        dim3 gProj(Dn / 128, Mn / 64, 1);
        gemm_wmma<64, 128, false, true, false, false><<<gProj, 256, 0, stream>>>(
            in_h, Dn, 0, 0, Wq_l, Dn, 0, 0, q, Dn, 0, 0, bq_l, nullptr, Dn, 1, 1.0f);
        gemm_wmma<64, 128, false, true, false, false><<<gProj, 256, 0, stream>>>(
            in_h, Dn, 0, 0, Wk_l, Dn, 0, 0, k, Dn, 0, 0, bk_l, nullptr, Dn, 1, 1.0f);
        gemm_wmma<64, 128, false, true, false, false><<<gProj, 256, 0, stream>>>(
            in_h, Dn, 0, 0, Wv_l, Dn, 0, 0, v, Dn, 0, 0, bv_l, nullptr, Dn, 1, 1.0f);

        // scores = (Q K^T)/8 written straight into d_out probs region
        dim3 gSc(Sn / 128, Sn / 64, Bn * Hn);
        gemm_wmma<64, 128, true, false, false, false><<<gSc, 256, 0, stream>>>(
            q, Dn, sQ_b, sQ_h, k, Dn, sQ_b, sQ_h,
            probs_l, Sn, sP_b, sP_h, nullptr, nullptr, HDn, Hn, 0.125f);

        softmax512<<<Bn * Hn * Sn, 256, 0, stream>>>(probs_l);

        // ctx = P @ V
        dim3 gCtx(HDn / 64, Sn / 128, Bn * Hn);
        gemm_wmma<128, 64, false, false, false, false><<<gCtx, 256, 0, stream>>>(
            probs_l, Sn, sP_b, sP_h, v, Dn, sQ_b, sQ_h,
            ctx, Dn, sQ_b, sQ_h, nullptr, nullptr, Sn, Hn, 1.0f);

        // h = ctx @ Wo + bo
        gemm_wmma<64, 128, false, true, false, false><<<gProj, 256, 0, stream>>>(
            ctx, Dn, 0, 0, Wo_l, Dn, 0, 0, h, Dn, 0, 0, bo_l, nullptr, Dn, 1, 1.0f);
    }

    // 3) attention_output = LN(att_op + x_embed)
    ln_kernel<<<Mn, 256, 0, stream>>>(h, xe, ln1_g, ln1_b, attln);

    // 4) hidden = relu(attention_output @ W1 + b1)
    gemm_wmma<64, 128, false, true, false, true><<<dim3(Fn / 128, Mn / 64, 1), 256, 0, stream>>>(
        attln, Dn, 0, 0, W1, Fn, 0, 0, hidden, Fn, 0, 0, b1, nullptr, Dn, 1, 1.0f);

    // 5) ffn_pre = hidden @ W2 + b2 + attention_output ; ffn = LN(ffn_pre)
    gemm_wmma<64, 128, false, true, true, false><<<dim3(Dn / 128, Mn / 64, 1), 256, 0, stream>>>(
        hidden, Fn, 0, 0, W2, Dn, 0, 0, ffn, Dn, 0, 0, b2, attln, Fn, 1, 1.0f);
    ln_kernel<<<Mn, 256, 0, stream>>>(ffn, nullptr, ln_ffn_g, ln_ffn_b, ffn);

    // 6) output = LN(ffn + attention_output) -> d_out
    ln_kernel<<<Mn, 256, 0, stream>>>(ffn, attln, ln2_g, ln2_b, outp);
}
